// Mamba_67980742362025
// MI455X (gfx1250) — matrix operand order repaired and checked by reading, not compiled, not run on hardware
//
#include <hip/hip_runtime.h>
#include <hip/hip_bf16.h>
#include <stdint.h>

#define D_MODEL 1024
#define D_INNER 2048
#define D_STATE 16
#define D_CONV  4
#define DT_RANK 64
#define BATCH   4
#define SEQLEN  2048
#define NROWS   (BATCH * SEQLEN)          // 8192 token rows
#define XPROJ_N (DT_RANK + 2 * D_STATE)   // 96

typedef __attribute__((ext_vector_type(16))) __bf16 v16bf;
typedef __attribute__((ext_vector_type(8)))  __bf16 bf16x8;
typedef __attribute__((ext_vector_type(8)))  float  v8f;
typedef __attribute__((ext_vector_type(4)))  int    v4i;

// ---------------------------------------------------------------------------
// Async global->LDS copy (CDNA5 ASYNCcnt path), with sync fallback.
// Builtin signature (from toolchain diagnostic): param0 = v4i addrspace(1)*.
// ---------------------------------------------------------------------------
#if __has_builtin(__builtin_amdgcn_global_load_async_to_lds_b128)
#define HAVE_ASYNC 1
#else
#define HAVE_ASYNC 0
#endif

#if HAVE_ASYNC
#if __has_builtin(__builtin_amdgcn_s_wait_asynccnt)
#define WAIT_ASYNC(n) __builtin_amdgcn_s_wait_asynccnt(n)
#else
#define WAIT_ASYNC(n) asm volatile("s_wait_asynccnt " #n ::: "memory")
#endif
__device__ __forceinline__ void cp16(const uint16_t* g, uint16_t* l) {
  __builtin_amdgcn_global_load_async_to_lds_b128(
      (__attribute__((address_space(1))) v4i*)(uintptr_t)g,
      (__attribute__((address_space(3))) v4i*)(uint32_t)(uintptr_t)l,
      0, 0);
}
#else
#define WAIT_ASYNC(n)
__device__ __forceinline__ void cp16(const uint16_t* g, uint16_t* l) {
  uint4 v = *(const uint4*)g;
  *(uint4*)l = v;
}
#endif

// ---------------------------------------------------------------------------
// fp32 -> bf16 (round-to-nearest-even), storage as u16 so host code never
// touches __bf16 ABI.
// ---------------------------------------------------------------------------
__device__ __forceinline__ uint16_t f2bf(float f) {
  union { float f; uint32_t u; } v; v.f = f;
  uint32_t r = v.u + 0x7FFFu + ((v.u >> 16) & 1u);
  return (uint16_t)(r >> 16);
}

__global__ void cvt_f32_bf16(const float* __restrict__ in,
                             uint16_t* __restrict__ out, int n) {
  int i = blockIdx.x * blockDim.x + threadIdx.x;
  if (i < n) out[i] = f2bf(in[i]);
}

// ---------------------------------------------------------------------------
// WMMA GEMM with double-buffered async LDS staging:
//   C[M,N] = act( A[M,K](bf16) * W[N,K]^T(bf16) + bias[N] )
// Block = 4 waves = 64(M) x 128(N) tile, K-step 32.
// Stage: A tile 64x32, B tile 128x32, rows padded to 40 elems (80B) so the
// 16-lane fragment reads land on 16 distinct bank groups (80/4=20, odd*4).
// Per stage each wave issues 6 async-copy instructions (512B/instr);
// WAIT_ASYNC(6) overlaps stage kt+1's DMA with stage kt's WMMAs.
// ---------------------------------------------------------------------------
#define ASTRIDE 40   // 32 + 8 pad (elements)

__global__ __launch_bounds__(128) void wmma_gemm_bf16(
    const uint16_t* __restrict__ A, int lda,
    const uint16_t* __restrict__ W, int ldw,
    float* __restrict__ C, int ldc,
    int M, int N, int K,
    const float* __restrict__ bias, int act) {
  __shared__ uint16_t sA[2][64 * ASTRIDE];
  __shared__ uint16_t sB[2][128 * ASTRIDE];

  const int tid  = threadIdx.x;
  const int lane = tid & 31;
  const int wave = tid >> 5;
  const int hi   = lane >> 4;     // half-wave select
  const int lm   = lane & 15;

  const int mblk = blockIdx.x * 64;
  const int nblk = blockIdx.y * 128;
  const int wm   = (wave & 1) * 32;   // wave M offset in tile
  const int wn   = (wave >> 1) * 64;  // wave N offset in tile

  const v8f vzero = {0.f, 0.f, 0.f, 0.f, 0.f, 0.f, 0.f, 0.f};
  v8f acc[2][4];
#pragma unroll
  for (int i = 0; i < 2; ++i)
#pragma unroll
    for (int j = 0; j < 4; ++j) acc[i][j] = vzero;

  // ---- stage copy: 256 A-chunks + 512 B-chunks of 16B, 128 threads ----
  auto stage_copy = [&](int s, int k0) {
#pragma unroll
    for (int t = 0; t < 2; ++t) {           // A: 64 rows x 4 chunks
      int c = tid + t * 128;
      int r = c >> 2, cc = c & 3;
      cp16(A + (size_t)(mblk + r) * lda + k0 + cc * 8,
           &sA[s][r * ASTRIDE + cc * 8]);
    }
#pragma unroll
    for (int t = 0; t < 4; ++t) {           // B: 128 rows x 4 chunks
      int c = tid + t * 128;
      int r = c >> 2, cc = c & 3;
      int n = nblk + r; if (n >= N) n = N - 1;  // clamp; store is guarded
      cp16(W + (size_t)n * ldw + k0 + cc * 8,
           &sB[s][r * ASTRIDE + cc * 8]);
    }
  };

  const int KT = K >> 5;
  stage_copy(0, 0);

  for (int kt = 0; kt < KT; ++kt) {
    const int s = kt & 1;
    if (kt + 1 < KT) {
      stage_copy(s ^ 1, (kt + 1) * 32);
      WAIT_ASYNC(6);     // previous stage's 6 per-wave copies complete
    } else {
      WAIT_ASYNC(0);
    }
    __syncthreads();     // stage kt visible to all waves

    v16bf a[2], b[4];
    // A fragment (16x32 bf16): elems 0..7 = K[hi*8..], 8..15 = K[16+hi*8..]
#pragma unroll
    for (int i = 0; i < 2; ++i) {
      const uint16_t* ap = &sA[s][(wm + i * 16 + lm) * ASTRIDE + hi * 8];
      ((bf16x8*)&a[i])[0] = *(const bf16x8*)(ap);
      ((bf16x8*)&a[i])[1] = *(const bf16x8*)(ap + 16);
    }
    // B fragment = W^T (32x16): lane holds row n, elems 0..7 = K[hi*8..], 8..15 = K[16+hi*8..]
#pragma unroll
    for (int j = 0; j < 4; ++j) {
      const uint16_t* bp = &sB[s][(wn + j * 16 + lm) * ASTRIDE + hi * 8];
      ((bf16x8*)&b[j])[0] = *(const bf16x8*)(bp);
      ((bf16x8*)&b[j])[1] = *(const bf16x8*)(bp + 16);
    }
#pragma unroll
    for (int i = 0; i < 2; ++i)
#pragma unroll
      for (int j = 0; j < 4; ++j)
        acc[i][j] = __builtin_amdgcn_wmma_f32_16x16x32_bf16(
            false, a[i], false, b[j], (short)0, acc[i][j], false, false);

    __syncthreads();     // all reads of stage kt done before it is reused
  }

  // C fragment layout: VGPR r -> (M = hi*8 + r, N = lm)
#pragma unroll
  for (int i = 0; i < 2; ++i) {
#pragma unroll
    for (int j = 0; j < 4; ++j) {
      int n = nblk + wn + j * 16 + lm;
      if (n >= N) continue;
      float bv = bias ? bias[n] : 0.f;
#pragma unroll
      for (int r = 0; r < 8; ++r) {
        int m = mblk + wm + i * 16 + hi * 8 + r;
        float v = acc[i][j][r] + bv;
        if (act == 1) v = (v > 20.f) ? v : __logf(1.f + __expf(v));  // softplus
        C[(size_t)m * ldc + n] = v;
      }
    }
  }
}

// ---------------------------------------------------------------------------
// Causal depthwise conv1d (taps=4) + SiLU.  xc is the left half of xz.
// Writes f32 (for scan) and bf16 (for x_proj GEMM).
// ---------------------------------------------------------------------------
__global__ void conv_silu_kernel(const float* __restrict__ xz,
                                 const float* __restrict__ conv_w,
                                 const float* __restrict__ conv_b,
                                 float* __restrict__ xact,
                                 uint16_t* __restrict__ xact_bf) {
  int idx = blockIdx.x * blockDim.x + threadIdx.x;
  if (idx >= NROWS * D_INNER) return;
  int d   = idx % D_INNER;
  int row = idx / D_INNER;
  int l   = row % SEQLEN;
  float acc = conv_b[d];
#pragma unroll
  for (int j = 0; j < D_CONV; ++j) {
    int ls = l - (D_CONV - 1) + j;
    if (ls >= 0)
      acc += conv_w[d * D_CONV + j] *
             xz[(size_t)(row - (D_CONV - 1) + j) * (2 * D_INNER) + d];
  }
  float s = acc / (1.f + __expf(-acc));  // SiLU
  xact[idx]    = s;
  xact_bf[idx] = f2bf(s);
}

// ---------------------------------------------------------------------------
// Selective scan: 4 lanes per (batch, channel), 4 states per lane.
// h recurrence is independent per state; per-step y is reduced with two
// shfl_xor hops.  32768 threads = 1024 waves (4x round-1 occupancy),
// 4 v_exp_f32 per lane-step.  Fuses D*u skip and SiLU(z) gate; emits bf16.
// ---------------------------------------------------------------------------
__global__ __launch_bounds__(256) void scan_kernel(
    const float* __restrict__ xz,      // [NROWS, 2*D_INNER] (z = cols 2048..)
    const float* __restrict__ xact,    // [NROWS, D_INNER]
    const float* __restrict__ delta,   // [NROWS, D_INNER]
    const float* __restrict__ x_dbl,   // [NROWS, 96] (B = 64..79, C = 80..95)
    const float* __restrict__ A_log,   // [D_INNER, 16]
    const float* __restrict__ Dp,      // [D_INNER]
    uint16_t* __restrict__ ybf) {      // [NROWS, D_INNER]
  const int sub = threadIdx.x & 3;           // state sub-group (4 states each)
  const int d   = blockIdx.x * 64 + (threadIdx.x >> 2);
  const int b   = blockIdx.y;
  const int n0  = sub * 4;

  float A[4];
#pragma unroll
  for (int j = 0; j < 4; ++j) A[j] = -__expf(A_log[d * D_STATE + n0 + j]);
  const float Dv = Dp[d];

  float h[4] = {0.f, 0.f, 0.f, 0.f};

  for (int l = 0; l < SEQLEN; ++l) {
    size_t row = (size_t)b * SEQLEN + l;
    float dt = delta[row * D_INNER + d];
    float u  = xact [row * D_INNER + d];
    float z  = xz   [row * (2 * D_INNER) + D_INNER + d];
    float4 Bv = *(const float4*)(x_dbl + row * XPROJ_N + DT_RANK + n0);
    float4 Cv = *(const float4*)(x_dbl + row * XPROJ_N + DT_RANK + D_STATE + n0);
    float du = dt * u;
    float y = 0.f;
#pragma unroll
    for (int j = 0; j < 4; ++j) {
      float dA = __expf(dt * A[j]);
      float bj = (j == 0) ? Bv.x : (j == 1) ? Bv.y : (j == 2) ? Bv.z : Bv.w;
      float cj = (j == 0) ? Cv.x : (j == 1) ? Cv.y : (j == 2) ? Cv.z : Cv.w;
      h[j] = h[j] * dA + du * bj;
      y += h[j] * cj;
    }
    y += __shfl_xor(y, 1, 32);
    y += __shfl_xor(y, 2, 32);
    if (sub == 0) {
      y += Dv * u;
      float gate = z / (1.f + __expf(-z));   // SiLU(z)
      ybf[row * D_INNER + d] = f2bf(y * gate);
    }
  }
}

// ---------------------------------------------------------------------------
// Host driver (graph-capture safe: only launches on `stream`).
// ---------------------------------------------------------------------------
extern "C" void kernel_launch(void* const* d_in, const int* in_sizes, int n_in,
                              void* d_out, int out_size, void* d_ws,
                              size_t ws_size, hipStream_t stream) {
  (void)in_sizes; (void)n_in; (void)out_size; (void)ws_size;
  const float* x         = (const float*)d_in[0];
  const float* in_proj_w = (const float*)d_in[1];
  const float* conv_w    = (const float*)d_in[2];
  const float* conv_b    = (const float*)d_in[3];
  const float* x_proj_w  = (const float*)d_in[4];
  const float* dt_proj_w = (const float*)d_in[5];
  const float* dt_proj_b = (const float*)d_in[6];
  const float* A_log     = (const float*)d_in[7];
  const float* Dp        = (const float*)d_in[8];
  const float* out_proj_w= (const float*)d_in[9];
  float* out = (float*)d_out;

  char* ws = (char*)d_ws;
  size_t off = 0;
  auto alloc = [&](size_t bytes) -> void* {
    void* p = ws + off;
    off = (off + bytes + 255) & ~(size_t)255;
    return p;
  };

  uint16_t* x_bf     = (uint16_t*)alloc((size_t)NROWS * D_MODEL * 2);
  uint16_t* w_in_bf  = (uint16_t*)alloc((size_t)2 * D_INNER * D_MODEL * 2);
  uint16_t* w_x_bf   = (uint16_t*)alloc((size_t)XPROJ_N * D_INNER * 2);
  uint16_t* w_dt_bf  = (uint16_t*)alloc((size_t)D_INNER * DT_RANK * 2);
  uint16_t* w_out_bf = (uint16_t*)alloc((size_t)D_MODEL * D_INNER * 2);
  float*    xz       = (float*)   alloc((size_t)NROWS * 2 * D_INNER * 4);
  float*    xact     = (float*)   alloc((size_t)NROWS * D_INNER * 4);
  uint16_t* xact_bf  = (uint16_t*)alloc((size_t)NROWS * D_INNER * 2);
  float*    x_dbl    = (float*)   alloc((size_t)NROWS * XPROJ_N * 4);
  uint16_t* x_dbl_bf = (uint16_t*)alloc((size_t)NROWS * XPROJ_N * 2);
  float*    delta    = (float*)   alloc((size_t)NROWS * D_INNER * 4);
  uint16_t* y_bf     = (uint16_t*)alloc((size_t)NROWS * D_INNER * 2);

  auto cdiv = [](int a, int b) { return (a + b - 1) / b; };
  auto cvt = [&](const float* src, uint16_t* dst, int n) {
    cvt_f32_bf16<<<cdiv(n, 256), 256, 0, stream>>>(src, dst, n);
  };

  // 0) one-time bf16 casts
  cvt(x,          x_bf,     NROWS * D_MODEL);
  cvt(in_proj_w,  w_in_bf,  2 * D_INNER * D_MODEL);
  cvt(x_proj_w,   w_x_bf,   XPROJ_N * D_INNER);
  cvt(dt_proj_w,  w_dt_bf,  D_INNER * DT_RANK);
  cvt(out_proj_w, w_out_bf, D_MODEL * D_INNER);

  // 1) xz = x @ in_proj_w^T           [8192, 4096]
  wmma_gemm_bf16<<<dim3(NROWS / 64, cdiv(2 * D_INNER, 128)), 128, 0, stream>>>(
      x_bf, D_MODEL, w_in_bf, D_MODEL, xz, 2 * D_INNER,
      NROWS, 2 * D_INNER, D_MODEL, nullptr, 0);

  // 2) causal depthwise conv + SiLU   -> xact (f32 + bf16)
  conv_silu_kernel<<<cdiv(NROWS * D_INNER, 256), 256, 0, stream>>>(
      xz, conv_w, conv_b, xact, xact_bf);

  // 3) x_dbl = xact @ x_proj_w^T      [8192, 96]
  wmma_gemm_bf16<<<dim3(NROWS / 64, cdiv(XPROJ_N, 128)), 128, 0, stream>>>(
      xact_bf, D_INNER, w_x_bf, D_INNER, x_dbl, XPROJ_N,
      NROWS, XPROJ_N, D_INNER, nullptr, 0);
  cvt(x_dbl, x_dbl_bf, NROWS * XPROJ_N);

  // 4) delta = softplus(x_dbl[:, :64] @ dt_proj_w^T + dt_proj_b)  [8192, 2048]
  wmma_gemm_bf16<<<dim3(NROWS / 64, cdiv(D_INNER, 128)), 128, 0, stream>>>(
      x_dbl_bf, XPROJ_N, w_dt_bf, DT_RANK, delta, D_INNER,
      NROWS, D_INNER, DT_RANK, dt_proj_b, 1);

  // 5) selective scan + D*u skip + SiLU(z) gate -> y_bf
  scan_kernel<<<dim3(D_INNER / 64, BATCH), 256, 0, stream>>>(
      xz, xact, delta, x_dbl, A_log, Dp, y_bf);

  // 6) out = y @ out_proj_w^T         [8192, 1024]
  wmma_gemm_bf16<<<dim3(NROWS / 64, cdiv(D_MODEL, 128)), 128, 0, stream>>>(
      y_bf, D_INNER, w_out_bf, D_INNER, out, D_MODEL,
      NROWS, D_MODEL, D_INNER, nullptr, 0);
}
